// LSTMModel_33681133535383
// MI455X (gfx1250) — matrix operand order repaired and checked by reading, not compiled, not run on hardware
//
#include <hip/hip_runtime.h>
#include <hip/hip_bf16.h>

#define HIDDEN 20
#define SEQ 24
#define NT 5              // 5 N-tiles of 16 cover the 80 gate outputs
#define WAVES 4
#define ROWS_PER_WAVE 16

typedef __attribute__((ext_vector_type(16))) __bf16 v16bf;
typedef __attribute__((ext_vector_type(8)))  float  v8f;

// ---- per-wave LDS layout (bytes) ----
// gates: 80 columns (col-major) x 20-dword stride  -> 6400 B (stride 20 => LDS-bank-conflict-free b128 ops)
// h:     16 rows x 80 B stride (32 bf16, K>=20 kept zero) -> 1280 B
// x:     16 rows x 24 f32 -> 1536 B
#define GATES_BYTES 6400
#define H_BYTES     1280
#define X_BYTES     1536
#define WAVE_LDS    (GATES_BYTES + H_BYTES + X_BYTES)   // 9216 B, 16B aligned

__device__ __forceinline__ void lds_fence() {
    // CDNA5 split dependency counters: wait on all outstanding LDS ops so
    // cross-lane LDS writes are visible before the next read phase.
    asm volatile("s_wait_dscnt 0" ::: "memory");
}

__device__ __forceinline__ float fsigmoid(float x) {
    // 1/(1+exp(-x)) with native v_exp_f32 (exp2) + v_rcp_f32
    return __builtin_amdgcn_rcpf(1.0f + __builtin_amdgcn_exp2f(-1.442695040888963f * x));
}
__device__ __forceinline__ float ftanh(float x) {
    // tanh(x) = 2*sigmoid(2x) - 1
    return __builtin_fmaf(2.0f, fsigmoid(2.0f * x), -1.0f);
}

__global__ __launch_bounds__(32 * WAVES)
void lstm_wmma_kernel(const float* __restrict__ x,     // [B, SEQ]
                      const float* __restrict__ W_ih,  // [80] (INPUT==1)
                      const float* __restrict__ W_hh,  // [80, 20]
                      const float* __restrict__ b_ih,  // [80]
                      const float* __restrict__ b_hh,  // [80]
                      const float* __restrict__ W_fc,  // [24, 20]
                      const float* __restrict__ b_fc,  // [24]
                      float* __restrict__ out,         // [B, SEQ]
                      long Btotal)
{
    __shared__ __align__(16) unsigned char smem[WAVES * WAVE_LDS];

    const int lane = threadIdx.x & 31;
    const int wave = threadIdx.x >> 5;
    const int col  = lane & 15;   // N / column index of WMMA fragments
    const int hi   = lane >> 4;   // lane half
    const long r0  = ((long)blockIdx.x * WAVES + wave) * ROWS_PER_WAVE;
    if (r0 + ROWS_PER_WAVE > Btotal) return;   // uniform per wave

    unsigned char* base      = smem + wave * WAVE_LDS;
    float*         gates_lds = (float*)base;                         // (colg,row) at dword colg*20+row
    unsigned char* h_lds     = base + GATES_BYTES;                   // row stride 80 B, bf16 at byte k*2
    float*         x_lds     = (float*)(base + GATES_BYTES + H_BYTES); // [16][24]

    // ---- load x tile: 384 contiguous floats, fully coalesced b128 ----
    {
        const float4* xg = (const float4*)(x + r0 * SEQ);
        float4*       xl = (float4*)x_lds;
        #pragma unroll
        for (int j = 0; j < 3; ++j) xl[j * 32 + lane] = xg[j * 32 + lane];
    }
    // ---- zero h_lds (h0 = 0, and K=20..31 padding stays zero forever) ----
    {
        float* hz = (float*)h_lds;
        #pragma unroll
        for (int j = 0; j < 10; ++j) hz[j * 32 + lane] = 0.0f;
    }

    // ---- W_hh B-fragments (bf16, K padded to 32), kept in VGPRs ----
    // B[K][N] = W_hh[16t + N][K]; lane holds N=col, K = hi*8 + e (e<8), 16 + hi*8 + (e-8) (e>=8)
    v16bf bh[NT];
    #pragma unroll
    for (int t = 0; t < NT; ++t) {
        const float* wrow = W_hh + (t * 16 + col) * HIDDEN;
        #pragma unroll
        for (int e = 0; e < 16; ++e) {
            const int K = hi * 8 + e + (e >= 8 ? 8 : 0);
            const float v = wrow[K < HIDDEN ? K : 0];
            bh[t][e] = (__bf16)(K < HIDDEN ? v : 0.0f);
        }
    }
    // ---- per-column bias and input weight (D-layout: depends only on N) ----
    float bias[NT], wih[NT];
    #pragma unroll
    for (int t = 0; t < NT; ++t) {
        bias[t] = b_ih[t * 16 + col] + b_hh[t * 16 + col];
        wih[t]  = W_ih[t * 16 + col];
    }

    // ---- cell state: 32 lanes x 10 = 320 = 16 rows x 20 hidden ----
    float c[10];
    #pragma unroll
    for (int q = 0; q < 10; ++q) c[q] = 0.0f;
    const int erow  = lane >> 1;        // elementwise row (0..15)
    const int ebase = (lane & 1) * 10;  // hidden-unit base (0 or 10)

    union AFrag { uint4 u[2]; v16bf v; };

    for (int ts = 0; ts < SEQ; ++ts) {
        lds_fence();
        // ---- A fragment of h_{ts-1}: 2 conflict-free ds_load_b128 per lane ----
        // 16-bit A 16x32 layout: lanes 0-15 hold K 0-7 & 16-23 of row M=col,
        // lanes 16-31 hold K 8-15 & 24-31 of the same row.
        AFrag af;
        const unsigned char* hrow = h_lds + (unsigned)(col * 80 + hi * 16);
        af.u[0] = *(const uint4*)(hrow);
        af.u[1] = *(const uint4*)(hrow + 32);

        // x values for this lane's 8 M-rows (LDS broadcast reads)
        float xv[8];
        #pragma unroll
        for (int v = 0; v < 8; ++v) xv[v] = x_lds[(v + 8 * hi) * SEQ + ts];

        #pragma unroll
        for (int t = 0; t < NT; ++t) {
            v8f acc;
            #pragma unroll
            for (int v = 0; v < 8; ++v) acc[v] = __builtin_fmaf(xv[v], wih[t], bias[t]);
            // gates tile (16 rows x 16 gate-cols), f32 accumulate
            acc = __builtin_amdgcn_wmma_f32_16x16x32_bf16(
                false, af.v, false, bh[t], (short)0, acc, false, false);
            // scatter to col-major gates LDS: two b128 stores, stride-20 => no bank conflicts
            float* gp = gates_lds + (t * 16 + col) * 20 + 8 * hi;
            *(float4*)(gp)     = make_float4(acc[0], acc[1], acc[2], acc[3]);
            *(float4*)(gp + 4) = make_float4(acc[4], acc[5], acc[6], acc[7]);
        }
        lds_fence();
        // ---- elementwise LSTM cell: 10 hidden units per lane ----
        #pragma unroll
        for (int q = 0; q < 10; ++q) {
            const int n = ebase + q;
            const float gi = gates_lds[(n     ) * 20 + erow];
            const float gf = gates_lds[(20 + n) * 20 + erow];
            const float gg = gates_lds[(40 + n) * 20 + erow];
            const float go = gates_lds[(60 + n) * 20 + erow];
            const float i = fsigmoid(gi);
            const float f = fsigmoid(gf);
            const float g = ftanh(gg);
            const float o = fsigmoid(go);
            c[q] = __builtin_fmaf(f, c[q], i * g);
            const float h = o * ftanh(c[q]);
            *(__bf16*)(h_lds + erow * 80 + n * 2) = (__bf16)h;   // K=n < 20
        }
    }

    // ================= final FC: out = hT @ W_fc.T + b_fc =================
    lds_fence();
    AFrag af;
    {
        const unsigned char* hrow = h_lds + (unsigned)(col * 80 + hi * 16);
        af.u[0] = *(const uint4*)(hrow);
        af.u[1] = *(const uint4*)(hrow + 32);
    }
    v16bf bf[2];
    #pragma unroll
    for (int ft = 0; ft < 2; ++ft) {
        const int s = ft * 16 + col;
        const float* wrow = W_fc + (s < SEQ ? s : 0) * HIDDEN;
        #pragma unroll
        for (int e = 0; e < 16; ++e) {
            const int K = hi * 8 + e + (e >= 8 ? 8 : 0);
            const float v = wrow[K < HIDDEN ? K : 0];
            bf[ft][e] = (__bf16)((s < SEQ && K < HIDDEN) ? v : 0.0f);
        }
    }
    v8f fc[2];
    #pragma unroll
    for (int ft = 0; ft < 2; ++ft) {
        const int s = ft * 16 + col;
        const float bb = (s < SEQ) ? b_fc[s < SEQ ? s : 0] : 0.0f;
        v8f acc;
        #pragma unroll
        for (int v = 0; v < 8; ++v) acc[v] = bb;
        fc[ft] = __builtin_amdgcn_wmma_f32_16x16x32_bf16(
            false, af.v, false, bf[ft], (short)0, acc, false, false);
    }
    // stage D tiles into LDS (reuse gates buffer) for coalesced output
    float* out_stage = gates_lds;   // flat [16][24]
    #pragma unroll
    for (int ft = 0; ft < 2; ++ft) {
        const int s = ft * 16 + col;
        if (s < SEQ) {
            #pragma unroll
            for (int v = 0; v < 8; ++v) out_stage[(v + 8 * hi) * SEQ + s] = fc[ft][v];
        }
    }
    lds_fence();
    {
        float4*       og = (float4*)(out + r0 * SEQ);
        const float4* ol = (const float4*)out_stage;
        #pragma unroll
        for (int j = 0; j < 3; ++j) og[j * 32 + lane] = ol[j * 32 + lane];
    }
}

extern "C" void kernel_launch(void* const* d_in, const int* in_sizes, int n_in,
                              void* d_out, int out_size, void* d_ws, size_t ws_size,
                              hipStream_t stream) {
    const float* x    = (const float*)d_in[0];
    const float* W_ih = (const float*)d_in[1];
    const float* W_hh = (const float*)d_in[2];
    const float* b_ih = (const float*)d_in[3];
    const float* b_hh = (const float*)d_in[4];
    const float* W_fc = (const float*)d_in[5];
    const float* b_fc = (const float*)d_in[6];
    float* out = (float*)d_out;

    const long B = (long)in_sizes[0] / SEQ;            // INPUT == 1
    const int rowsPerBlock = WAVES * ROWS_PER_WAVE;    // 64
    const int grid = (int)((B + rowsPerBlock - 1) / rowsPerBlock);

    lstm_wmma_kernel<<<grid, 32 * WAVES, 0, stream>>>(
        x, W_ih, W_hh, b_ih, b_hh, W_fc, b_fc, out, B);
}
